// NegativeSampling_91207925498217
// MI455X (gfx1250) — matrix operand order repaired and checked by reading, not compiled, not run on hardware
//
#include <hip/hip_runtime.h>
#include <hip/hip_bf16.h>

// ---- problem constants (from reference) ----
#define VOCAB     50000
#define HDIM      128
#define NROWS     2048          // B*C*S = 128*2*8
#define HALF_ROWS 1024
#define NEG       5
#define SLOTS     6             // 1 positive + 5 negatives
#define NPAIR     (NROWS*SLOTS) // 12288
#define NTILES    (NPAIR/16)    // 768 WMMA tiles
#define CHUNK     4096          // log_dist LDS tile (16KB)
#define HALF_CTR  51200000u     // NROWS*VOCAB/2 (JAX threefry counter split)
#define TINYF     1.17549435e-38f

typedef __attribute__((ext_vector_type(16))) _Float16     v16h;
typedef __attribute__((ext_vector_type(8)))  float        v8f;
typedef __attribute__((ext_vector_type(4)))  unsigned int u32x4;
typedef __attribute__((ext_vector_type(8)))  int          i32x8;
typedef __attribute__((ext_vector_type(4)))  int          i32x4;

#if defined(__has_builtin)
#if __has_builtin(__builtin_amdgcn_tensor_load_to_lds) && \
    __has_builtin(__builtin_amdgcn_s_wait_tensorcnt)
#define USE_TDM 1
#endif
#endif

// ----------------- threefry2x32 (exact JAX PRNG core) -----------------
__device__ __forceinline__ unsigned rotl32(unsigned x, unsigned n) {
  return (x << n) | (x >> (32u - n));
}
__device__ __forceinline__ void threefry2x32(unsigned k0, unsigned k1,
                                             unsigned x0, unsigned x1,
                                             unsigned& o0, unsigned& o1) {
  const unsigned k2 = k0 ^ k1 ^ 0x1BD11BDAu;
  x0 += k0; x1 += k1;
#define TFR(r) { x0 += x1; x1 = rotl32(x1,(r)); x1 ^= x0; }
  TFR(13) TFR(15) TFR(26) TFR(6)  x0 += k1; x1 += k2 + 1u;
  TFR(17) TFR(29) TFR(16) TFR(24) x0 += k2; x1 += k0 + 2u;
  TFR(13) TFR(15) TFR(26) TFR(6)  x0 += k0; x1 += k1 + 3u;
  TFR(17) TFR(29) TFR(16) TFR(24) x0 += k1; x1 += k2 + 4u;
  TFR(13) TFR(15) TFR(26) TFR(6)  x0 += k2; x1 += k0 + 5u;
#undef TFR
  o0 = x0; o1 = x1;
}

__device__ __forceinline__ float gumbel_from_bits(unsigned bits) {
  // JAX: u = uniform(tiny, 1); g = -log(-log(u))
  float f = __uint_as_float((bits >> 9) | 0x3f800000u) - 1.0f; // [0,1)
  float u = f * (1.0f - TINYF) + TINYF;
  u = fmaxf(u, TINYF);
  return -__logf(-__logf(u));
}

// ----------------- per-thread top-5 (descending) -----------------
__device__ __forceinline__ void topk_insert(float sc, int v,
                                            float (&s)[NEG], int (&id)[NEG]) {
  if (sc <= s[NEG - 1]) return;
  s[NEG - 1] = sc; id[NEG - 1] = v;
#pragma unroll
  for (int k = NEG - 1; k > 0; --k) {
    if (s[k] > s[k - 1]) {
      float ts = s[k]; s[k] = s[k - 1]; s[k - 1] = ts;
      int   ti = id[k]; id[k] = id[k - 1]; id[k - 1] = ti;
    }
  }
}

// ----------------- TDM descriptor build + issue (1-D tile) -----------------
#ifdef USE_TDM
__device__ __forceinline__ unsigned lds_addr_of(const void* p) {
  // flat LDS aperture: low 32 bits are the LDS byte offset
  return (unsigned)(unsigned long long)p;
}
__device__ __forceinline__ void tdm_issue_1d(const float* src, unsigned ldsByteAddr,
                                             int nElems) {
  unsigned long long ga = (unsigned long long)src;
  u32x4 g0;
  g0[0] = 1u;                                   // count=1, user descriptor
  g0[1] = ldsByteAddr;                          // lds_addr
  g0[2] = (unsigned)(ga & 0xffffffffu);         // global_addr[31:0]
  g0[3] = (unsigned)((ga >> 32) & 0x01ffffffu)  // global_addr[56:32]
        | (2u << 30);                           // type=2 ("image")
  unsigned td0 = (unsigned)nElems;
  i32x8 g1;
  g1[0] = (int)(2u << 16);                      // workgroup_mask=0, data_size=4B
  g1[1] = (int)((td0 & 0xffffu) << 16);         // tensor_dim0[15:0]
  g1[2] = (int)(((td0 >> 16) & 0xffffu)         // tensor_dim0[31:16]
        | (1u << 16));                          // tensor_dim1=1 (low 16)
  g1[3] = (int)((td0 & 0xffffu) << 16);         // tile_dim0 = nElems
  g1[4] = 0;                                    // tile_dim1=0, tile_dim2=0 (1-D)
  g1[5] = (int)td0;                             // tensor_dim0_stride[31:0]
  g1[6] = 0;
  g1[7] = 0;
  i32x4 z4;
  z4[0] = 0; z4[1] = 0; z4[2] = 0; z4[3] = 0;
  i32x8 z8;
#pragma unroll
  for (int i = 0; i < 8; ++i) z8[i] = 0;
  // 6-arg variant (clang-23 / therock-10.0 headers)
  __builtin_amdgcn_tensor_load_to_lds(g0, g1, z4, z4, z8, 0);
}
#endif

// ----------------- kernel 1: S = sum(freq^0.75), deterministic -----------------
__global__ void __launch_bounds__(256) k_noise_norm(const float* __restrict__ freqs,
                                                    float* __restrict__ sumOut) {
  __shared__ float red[256];
  float acc = 0.0f;
  for (int v = threadIdx.x; v < VOCAB; v += 256) acc += __powf(freqs[v], 0.75f);
  red[threadIdx.x] = acc;
  __syncthreads();
  for (int s = 128; s >= 1; s >>= 1) {
    if ((int)threadIdx.x < s) red[threadIdx.x] += red[threadIdx.x + s];
    __syncthreads();
  }
  if (threadIdx.x == 0) sumOut[0] = red[0];
}

// ----------------- kernel 2: log_dist[v] = 0.75*log f - log S -----------------
__global__ void __launch_bounds__(256) k_logdist(const float* __restrict__ freqs,
                                                 const float* __restrict__ sumP,
                                                 float* __restrict__ logd) {
  int v = blockIdx.x * 256 + threadIdx.x;
  if (v < VOCAB) logd[v] = 0.75f * __logf(freqs[v]) - __logf(sumP[0]);
}

// ----------------- chunk processing (fused gumbel + top-5) -----------------
__device__ __forceinline__ void process_chunk(const float* __restrict__ ld,
                                              int base, int limit, unsigned pbase,
                                              int tA, int tB,
                                              float (&sA)[NEG], int (&iA)[NEG],
                                              float (&sB)[NEG], int (&iB)[NEG]) {
  for (int j = threadIdx.x; j < limit; j += 256) {
    const int v = base + j;
    unsigned y0, y1;
    // JAX counter layout: pair p = row*V+v encrypts elements p (row) and p+half (row+1024)
    threefry2x32(0u, 42u, pbase + (unsigned)v, pbase + (unsigned)v + HALF_CTR, y0, y1);
    const float ldv = ld[j];
    if (v != tA) topk_insert(ldv + gumbel_from_bits(y0), v, sA, iA);
    if (v != tB) topk_insert(ldv + gumbel_from_bits(y1), v, sB, iB);
  }
}

// ----------------- block-wide top-5 tree merge -----------------
__device__ __forceinline__ void block_top5(float (&s)[NEG], int (&id)[NEG],
                                           float* s_rs, int* s_ri,
                                           int* __restrict__ outIdx) {
  const int tid = threadIdx.x;
#pragma unroll
  for (int k = 0; k < NEG; ++k) { s_rs[tid * NEG + k] = s[k]; s_ri[tid * NEG + k] = id[k]; }
  __syncthreads();
  for (int st = 128; st >= 1; st >>= 1) {
    if (tid < st) {
      float as[NEG], bs[NEG]; int ai[NEG], bi[NEG];
#pragma unroll
      for (int k = 0; k < NEG; ++k) {
        as[k] = s_rs[tid * NEG + k];        ai[k] = s_ri[tid * NEG + k];
        bs[k] = s_rs[(tid + st) * NEG + k]; bi[k] = s_ri[(tid + st) * NEG + k];
      }
      float ms[NEG]; int mi[NEG]; int ia = 0, ib = 0;
#pragma unroll
      for (int k = 0; k < NEG; ++k) {
        bool takeA = (as[ia] >= bs[ib]);
        ms[k] = takeA ? as[ia] : bs[ib];
        mi[k] = takeA ? ai[ia] : bi[ib];
        ia += takeA ? 1 : 0; ib += takeA ? 0 : 1;
      }
#pragma unroll
      for (int k = 0; k < NEG; ++k) { s_rs[tid * NEG + k] = ms[k]; s_ri[tid * NEG + k] = mi[k]; }
    }
    __syncthreads();
  }
  if (tid == 0) {
#pragma unroll
    for (int k = 0; k < NEG; ++k) outIdx[k] = s_ri[k];
  }
  __syncthreads();
}

// ----------------- kernel 3: fused gumbel-top-k sampling -----------------
// block i handles rows i and i+1024 (exact JAX threefry counter split)
__global__ void __launch_bounds__(256) k_sample(const float* __restrict__ logd,
                                                const int* __restrict__ tgt,
                                                int* __restrict__ negIdx) {
  __shared__ float s_ld[2][CHUNK];          // 32KB: double-buffered log_dist tiles
  __shared__ float s_rs[256 * NEG];
  __shared__ int   s_ri[256 * NEG];

  const int rA = blockIdx.x;
  const int rB = rA + HALF_ROWS;
  const int tA = tgt[rA];
  const int tB = tgt[rB];
  float sA[NEG], sB[NEG]; int iA[NEG], iB[NEG];
#pragma unroll
  for (int k = 0; k < NEG; ++k) { sA[k] = -3.0e38f; sB[k] = -3.0e38f; iA[k] = -1; iB[k] = -1; }

  const unsigned pbase = (unsigned)rA * (unsigned)VOCAB;
  const int nChunks = (VOCAB + CHUNK - 1) / CHUNK;
  const bool wave0 = (threadIdx.x < 32);

#ifdef USE_TDM
  if (wave0) tdm_issue_1d(logd, lds_addr_of(&s_ld[0][0]), CHUNK);
  for (int c = 0; c < nChunks; ++c) {
    if (wave0) __builtin_amdgcn_s_wait_tensorcnt(0);   // chunk c landed in LDS
    __syncthreads();
    if (c + 1 < nChunks && wave0) {                    // prefetch chunk c+1
      int nb = VOCAB - (c + 1) * CHUNK; if (nb > CHUNK) nb = CHUNK;
      tdm_issue_1d(logd + (c + 1) * CHUNK, lds_addr_of(&s_ld[(c + 1) & 1][0]), nb);
    }
    const int base = c * CHUNK;
    int limit = VOCAB - base; if (limit > CHUNK) limit = CHUNK;
    process_chunk(s_ld[c & 1], base, limit, pbase, tA, tB, sA, iA, sB, iB);
    __syncthreads();                                   // done reading buf before reuse
  }
#else
  for (int c = 0; c < nChunks; ++c) {
    const int base = c * CHUNK;
    int limit = VOCAB - base; if (limit > CHUNK) limit = CHUNK;
    __syncthreads();
    for (int j = threadIdx.x; j < limit; j += 256) s_ld[0][j] = logd[base + j];
    __syncthreads();
    process_chunk(s_ld[0], base, limit, pbase, tA, tB, sA, iA, sB, iB);
  }
#endif

  block_top5(sA, iA, s_rs, s_ri, negIdx + (size_t)rA * NEG);
  block_top5(sB, iB, s_rs, s_ri, negIdx + (size_t)rB * NEG);
}

// ----------------- kernel 4: gathered dot products via WMMA -----------------
// 16 (embedding-row, fc-row) pairs per wave: A=16x32 f16, B=32x16 f16, 4 k-steps,
// diagonal of the 16x16 f32 accumulator = the 16 logits.
__global__ void __launch_bounds__(256) k_logits(const float* __restrict__ emb,
                                                const float* __restrict__ fc,
                                                const int* __restrict__ tgt,
                                                const int* __restrict__ negIdx,
                                                float* __restrict__ logits) {
  const int lane = threadIdx.x & 31;
  const int tile = (blockIdx.x * blockDim.x + threadIdx.x) >> 5;   // grid exact: 768 waves
  const int m = lane & 15;
  const int h = lane >> 4;
  const int q = tile * 16 + m;         // pair id
  const int r = q / SLOTS;
  const int t = q - r * SLOTS;
  const int fidx = (t == 0) ? tgt[r] : negIdx[r * NEG + (t - 1)];
  const float* erow = emb + (size_t)r * HDIM;
  const float* frow = fc + (size_t)fidx * HDIM;
  __builtin_prefetch(frow, 0, 0);      // global_prefetch_b8 for the gathered row

  v8f acc;
#pragma unroll
  for (int i = 0; i < 8; ++i) acc[i] = 0.0f;

#pragma unroll
  for (int kk = 0; kk < 4; ++kk) {
    const int kb = kk * 32;
    v16h a, b;
    // A 16x32 f16 layout: lane (M=m, half h) holds K = [h*8, h*8+8) U [16+h*8, 16+h*8+8)
#pragma unroll
    for (int i = 0; i < 8; ++i) a[i]     = (_Float16)erow[kb + h * 8 + i];
#pragma unroll
    for (int i = 0; i < 8; ++i) a[8 + i] = (_Float16)erow[kb + 16 + h * 8 + i];
    // B 32x16 f16 layout: lane (N=m, half h) holds K = [h*8, h*8+8) U [16+h*8, 16+h*8+8)
#pragma unroll
    for (int i = 0; i < 8; ++i) b[i]     = (_Float16)frow[kb + h * 8 + i];
#pragma unroll
    for (int i = 0; i < 8; ++i) b[8 + i] = (_Float16)frow[kb + 16 + h * 8 + i];
    acc = __builtin_amdgcn_wmma_f32_16x16x32_f16(false, a, false, b,
                                                 (short)0, acc, false, false);
  }
  // diagonal of C: (d,d) d<8 -> lane d, acc[d]; d>=8 -> lane d+16, acc[d-8]
  const int sel = (lane < 16) ? lane : (lane - 24);
  float diag = 0.0f;
#pragma unroll
  for (int i = 0; i < 8; ++i) if (sel == i) diag = acc[i];
  if (lane < 8)        logits[tile * 16 + lane]        = diag;
  else if (lane >= 24) logits[tile * 16 + (lane - 16)] = diag;
}

// ----------------- kernel 5: deterministic loss reduction -----------------
__global__ void __launch_bounds__(256) k_loss(const float* __restrict__ logits,
                                              float* __restrict__ out) {
  __shared__ float red[256];
  float acc = 0.0f;
  for (int i = threadIdx.x; i < NPAIR; i += 256) {
    float x = logits[i];
    int t = i % SLOTS;
    if (t == 0) {
      float y = -x;
      acc += fmaxf(y, 0.0f) + log1pf(__expf(-fabsf(y)));          // softplus(-pos)
    } else {
      acc += (fmaxf(x, 0.0f) + log1pf(__expf(-fabsf(x)))) * (1.0f / 128.0f);
    }
  }
  red[threadIdx.x] = acc;
  __syncthreads();
  for (int s = 128; s >= 1; s >>= 1) {
    if ((int)threadIdx.x < s) red[threadIdx.x] += red[threadIdx.x + s];
    __syncthreads();
  }
  if (threadIdx.x == 0) out[0] = red[0];
}

// ----------------- launch -----------------
extern "C" void kernel_launch(void* const* d_in, const int* in_sizes, int n_in,
                              void* d_out, int out_size, void* d_ws, size_t ws_size,
                              hipStream_t stream) {
  const float* emb   = (const float*)d_in[0];   // (128,2,8,128)
  const int*   tgt   = (const int*)d_in[1];     // (128,2,8)
  const float* fc    = (const float*)d_in[2];   // (50000,128)
  const float* freqs = (const float*)d_in[3];   // (50000,)

  float* ws_f   = (float*)d_ws;
  float* logd   = ws_f;                                   // VOCAB floats
  float* sumP   = ws_f + VOCAB;                           // 1 float
  int*   negIdx = (int*)(ws_f + VOCAB + 1);               // NROWS*NEG ints
  float* logits = (float*)(negIdx + (size_t)NROWS * NEG); // NPAIR floats
  float* out    = (float*)d_out;

  k_noise_norm<<<1, 256, 0, stream>>>(freqs, sumP);
  k_logdist<<<(VOCAB + 255) / 256, 256, 0, stream>>>(freqs, sumP, logd);
  k_sample<<<HALF_ROWS, 256, 0, stream>>>(logd, tgt, negIdx);
  k_logits<<<NTILES / 8, 256, 0, stream>>>(emb, fc, tgt, negIdx, logits);
  k_loss<<<1, 256, 0, stream>>>(logits, out);
}
